// Memory_36550171689427
// MI455X (gfx1250) — matrix operand order repaired and checked by reading, not compiled, not run on hardware
//
#include <hip/hip_runtime.h>

// ---------------------------------------------------------------------------
// Memory-attention for MI455X (gfx1250, wave32).
//   B=8 S=512 D=256 H=4 M=20 VOCAB=50000, num = B*S = 4096
// Plan:
//   1) prep_weights : Wq/Wk/Wv (f32, [k][c]) -> bf16 transposed Wt[c][k] in ws
//   2) gemm256_bf16 : rows @ Wt + bias -> bf16  (WMMA f32_16x16x32_bf16)
//        B fragments register-resident; A tile staged in LDS (double buffer).
//   3) attn_kernel  : one wave per token; gather 20 bf16 K/V rows, 4-head
//        softmax attention in fp32, defaults for fully-masked rows.
// ---------------------------------------------------------------------------

typedef __bf16 bf16_t;
typedef __attribute__((ext_vector_type(16))) __bf16 v16bf;
typedef __attribute__((ext_vector_type(8)))  __bf16 v8bf;
typedef __attribute__((ext_vector_type(8)))  float  v8f;

#define DIM    256
#define MSLOT  20
#define APITCH 264   // LDS row pitch in bf16 (16B-aligned, conflict-friendly)

// ---------------------------------------------------------------------------
// Kernel 1: transpose + convert the three 256x256 weights to bf16 Wt[c][k].
// ---------------------------------------------------------------------------
__global__ void prep_weights(const float* __restrict__ Wq,
                             const float* __restrict__ Wk,
                             const float* __restrict__ Wv,
                             bf16_t* __restrict__ WtQ,
                             bf16_t* __restrict__ WtK,
                             bf16_t* __restrict__ WtV) {
    int i = blockIdx.x * blockDim.x + threadIdx.x;
    if (i >= DIM * DIM) return;
    int k = i >> 8;        // source row    (input-feature index)
    int c = i & 255;       // source column (output-feature index)
    int t = c * DIM + k;   // transposed position
    WtQ[t] = (bf16_t)Wq[i];
    WtK[t] = (bf16_t)Wk[i];
    WtV[t] = (bf16_t)Wv[i];
}

// ---------------------------------------------------------------------------
// Kernel 2: Out[R][256](bf16) = A[R][256](f32) @ W + bias, with W given as
// bf16 Wt[c][k] (pre-transposed).
//
// Block: 256 threads = 8 waves, 128 rows (8 row tiles of 16), all 256 cols.
// Wave w owns column tiles w and w+8; preloads ALL its B fragments for the
// whole K range into registers (16 x v16bf = 128 VGPRs) -- weights are tiny
// and constant, activations are the 51 MB stream.
// Per row tile: cooperative stage of A (f32 -> bf16) into LDS (double
// buffered), then each wave does 8 k-steps of ds_load A + 2 WMMAs.
//
// Fragment addressing (contiguous k, same order for A and B):
//   A (16x32 bf16): lane<16: row=lane,    K = k0..k0+15
//                   lane>=16: row=lane-16, +16.
//   B (32x16 bf16): lane<16: col=lane,    K = k0..k0+15 ; lane>=16: +16.
//   C/D (16x16 f32): VGPR j -> row j (lanes 0-15) / j+8 (lanes 16-31),
//                    col = lane&15.
// ---------------------------------------------------------------------------
__global__ __launch_bounds__(256) void
gemm256_bf16(const float*  __restrict__ A,
             const bf16_t* __restrict__ Wt,
             const float*  __restrict__ bias,
             bf16_t*       __restrict__ Out,
             int R) {
    __shared__ bf16_t lds[2][16 * APITCH];

    const int tid    = threadIdx.x;
    const int lane   = tid & 31;
    const int wave   = tid >> 5;
    const int half   = lane >> 4;          // 0 or 1
    const int lanelo = lane & 15;
    const int row0   = blockIdx.x * 128;

    // ---- preload B fragments for this wave's two column tiles (resident)
    v16bf bfr0[8], bfr1[8];
    #pragma unroll
    for (int ks = 0; ks < 8; ++ks) {
        const bf16_t* bp0 = Wt + (size_t)(wave * 16 + lanelo) * DIM
                               + ks * 32 + (half << 4);
        const bf16_t* bp1 = Wt + (size_t)((wave + 8) * 16 + lanelo) * DIM
                               + ks * 32 + (half << 4);
        v8bf x0 = *(const v8bf*)(bp0 + 0);
        v8bf x1 = *(const v8bf*)(bp0 + 8);
        bfr0[ks] = __builtin_shufflevector(x0, x1,
                     0,1,2,3,4,5,6,7,8,9,10,11,12,13,14,15);
        v8bf y0 = *(const v8bf*)(bp1 + 0);
        v8bf y1 = *(const v8bf*)(bp1 + 8);
        bfr1[ks] = __builtin_shufflevector(y0, y1,
                     0,1,2,3,4,5,6,7,8,9,10,11,12,13,14,15);
    }

    const float bias0 = bias[wave * 16 + lanelo];
    const float bias1 = bias[(wave + 8) * 16 + lanelo];

    // staging role: thread t loads row (t>>4), 16-float segment (t&15)
    const int srow = tid >> 4;
    const int sseg = tid & 15;

    for (int rt = 0; rt < 8; ++rt) {
        const int rowbase = row0 + rt * 16;

        // ---- stage A tile (16 x 256 f32 -> bf16) into lds[rt&1]
        {
            int r = rowbase + srow;
            if (r >= R) r = R - 1;                    // clamp (stores guarded)
            const float* ap = A + (size_t)r * DIM + sseg * 16;
            float4 f0 = *(const float4*)(ap +  0);
            float4 f1 = *(const float4*)(ap +  4);
            float4 f2 = *(const float4*)(ap +  8);
            float4 f3 = *(const float4*)(ap + 12);
            v8bf p0, p1;
            p0[0]=(bf16_t)f0.x; p0[1]=(bf16_t)f0.y; p0[2]=(bf16_t)f0.z; p0[3]=(bf16_t)f0.w;
            p0[4]=(bf16_t)f1.x; p0[5]=(bf16_t)f1.y; p0[6]=(bf16_t)f1.z; p0[7]=(bf16_t)f1.w;
            p1[0]=(bf16_t)f2.x; p1[1]=(bf16_t)f2.y; p1[2]=(bf16_t)f2.z; p1[3]=(bf16_t)f2.w;
            p1[4]=(bf16_t)f3.x; p1[5]=(bf16_t)f3.y; p1[6]=(bf16_t)f3.z; p1[7]=(bf16_t)f3.w;
            bf16_t* dst = &lds[rt & 1][srow * APITCH + sseg * 16];
            *(v8bf*)(dst + 0) = p0;
            *(v8bf*)(dst + 8) = p1;

            if (rt < 7) {                              // prefetch next tile
                int rn = rowbase + 16 + srow;
                if (rn >= R) rn = R - 1;
                __builtin_prefetch(A + (size_t)rn * DIM + sseg * 16, 0, 1);
            }
        }
        __syncthreads();

        // ---- compute: 8 k-steps, A from LDS, B resident
        v8f acc0 = {}, acc1 = {};
        const bf16_t* abase = &lds[rt & 1][lanelo * APITCH];
        #pragma unroll
        for (int ks = 0; ks < 8; ++ks) {
            const bf16_t* ap0 = abase + ks * 32 + (half << 4);
            v8bf a0 = *(const v8bf*)(ap0 +  0);
            v8bf a1 = *(const v8bf*)(ap0 +  8);
            v16bf a = __builtin_shufflevector(a0, a1,
                        0,1,2,3,4,5,6,7,8,9,10,11,12,13,14,15);
            acc0 = __builtin_amdgcn_wmma_f32_16x16x32_bf16(
                       false, a, false, bfr0[ks], (short)0, acc0, false, false);
            acc1 = __builtin_amdgcn_wmma_f32_16x16x32_bf16(
                       false, a, false, bfr1[ks], (short)0, acc1, false, false);
        }

        // ---- epilogue: + bias, convert, store (guarded)
        const int col0 = wave * 16 + lanelo;
        const int col1 = (wave + 8) * 16 + lanelo;
        #pragma unroll
        for (int j = 0; j < 8; ++j) {
            int row = rowbase + j + (half << 3);
            if (row < R) {
                Out[(size_t)row * DIM + col0] = (bf16_t)(acc0[j] + bias0);
                Out[(size_t)row * DIM + col1] = (bf16_t)(acc1[j] + bias1);
            }
        }
        // double-buffered LDS: single barrier per iteration is sufficient
    }
}

// ---------------------------------------------------------------------------
// Kernel 3: attention.  One wave per token.  lane owns 8 channels
// (ch0 = lane*8); 8 lanes per head (dk=64).  Cross-lane dot reduction via
// shfl_xor 1/2/4 inside the head group.
// ---------------------------------------------------------------------------
__global__ __launch_bounds__(256) void
attn_kernel(const bf16_t* __restrict__ Qb,
            const bf16_t* __restrict__ Kb,
            const bf16_t* __restrict__ Vb,
            const int*    __restrict__ idx,
            const float*  __restrict__ def_h,
            const float*  __restrict__ def_l,
            float*        __restrict__ outh,
            float*        __restrict__ outl,
            int num) {
    const int lane = threadIdx.x & 31;
    const int n    = (int)((blockIdx.x * blockDim.x + threadIdx.x) >> 5);
    if (n >= num) return;
    const int ch0 = lane * 8;

    // query chunk (bf16 -> f32)
    v8bf qv = *(const v8bf*)(Qb + (size_t)n * DIM + ch0);
    float qf[8];
    #pragma unroll
    for (int j = 0; j < 8; ++j) qf[j] = (float)qv[j];

    int  idv[MSLOT];
    v8bf kf[MSLOT], vf[MSLOT];
    float sc[MSLOT];

    #pragma unroll
    for (int m = 0; m < MSLOT; ++m) {
        int id = idx[(size_t)n * MSLOT + m];
        idv[m] = id;
        kf[m] = *(const v8bf*)(Kb + (size_t)id * DIM + ch0);
        vf[m] = *(const v8bf*)(Vb + (size_t)id * DIM + ch0);
        float s = 0.f;
        #pragma unroll
        for (int j = 0; j < 8; ++j) s += qf[j] * (float)kf[m][j];
        s += __shfl_xor(s, 1, 32);
        s += __shfl_xor(s, 2, 32);
        s += __shfl_xor(s, 4, 32);       // reduced over the 8-lane head group
        sc[m] = s * 0.125f;              // / sqrt(dk) = / 8
    }

    // masked softmax over the 20 slots
    float mx = -3.0e38f;
    #pragma unroll
    for (int m = 0; m < MSLOT; ++m)
        if (idv[m] != 0) mx = fmaxf(mx, sc[m]);

    float p[MSLOT];
    float den = 0.f;
    #pragma unroll
    for (int m = 0; m < MSLOT; ++m) {
        p[m] = (idv[m] != 0) ? __expf(sc[m] - mx) : 0.f;
        den += p[m];
    }
    const bool empty = (den == 0.f);     // fully-masked row
    const float rden = empty ? 0.f : 1.f / den;

    float ah[8] = {}, al[8] = {};
    #pragma unroll
    for (int m = 0; m < MSLOT; ++m) {
        float pm = p[m] * rden;
        #pragma unroll
        for (int j = 0; j < 8; ++j) {
            ah[j] += pm * (float)kf[m][j];
            al[j] += pm * (float)vf[m][j];
        }
    }
    if (empty) {
        #pragma unroll
        for (int j = 0; j < 8; ++j) {
            ah[j] = def_h[ch0 + j];
            al[j] = def_l[ch0 + j];
        }
    }

    float4* ph = (float4*)(outh + (size_t)n * DIM + ch0);
    float4* pl = (float4*)(outl + (size_t)n * DIM + ch0);
    ph[0] = make_float4(ah[0], ah[1], ah[2], ah[3]);
    ph[1] = make_float4(ah[4], ah[5], ah[6], ah[7]);
    pl[0] = make_float4(al[0], al[1], al[2], al[3]);
    pl[1] = make_float4(al[4], al[5], al[6], al[7]);
}

// ---------------------------------------------------------------------------
// Launcher.  Input order: query_h, idx, mem_h, mem_l, default_h, default_l,
//                         Wq, bq, Wk, bk, Wv, bv
// d_out = [h (num*256 f32) | l (num*256 f32)]
// ---------------------------------------------------------------------------
extern "C" void kernel_launch(void* const* d_in, const int* in_sizes, int n_in,
                              void* d_out, int out_size, void* d_ws, size_t ws_size,
                              hipStream_t stream) {
    const float* query_h = (const float*)d_in[0];
    const int*   idx     = (const int*)  d_in[1];
    const float* mem_h   = (const float*)d_in[2];
    const float* mem_l   = (const float*)d_in[3];
    const float* def_h   = (const float*)d_in[4];
    const float* def_l   = (const float*)d_in[5];
    const float* Wq      = (const float*)d_in[6];
    const float* bq      = (const float*)d_in[7];
    const float* Wk      = (const float*)d_in[8];
    const float* bk      = (const float*)d_in[9];
    const float* Wv      = (const float*)d_in[10];
    const float* bv      = (const float*)d_in[11];

    const int num   = in_sizes[0] / DIM;   // 4096
    const int vocab = in_sizes[2] / DIM;   // 50000

    // workspace layout (bf16)
    bf16_t* WtQ = (bf16_t*)d_ws;
    bf16_t* WtK = WtQ + DIM * DIM;
    bf16_t* WtV = WtK + DIM * DIM;
    bf16_t* Qb  = WtV + DIM * DIM;
    bf16_t* Kb  = Qb + (size_t)num   * DIM;
    bf16_t* Vb  = Kb + (size_t)vocab * DIM;

    // 1) weight transpose + bf16 convert
    prep_weights<<<(DIM * DIM + 255) / 256, 256, 0, stream>>>(
        Wq, Wk, Wv, WtQ, WtK, WtV);

    // 2) WMMA projections (Q over tokens; K/V over the whole vocab)
    gemm256_bf16<<<(num   + 127) / 128, 256, 0, stream>>>(query_h, WtQ, bq, Qb, num);
    gemm256_bf16<<<(vocab + 127) / 128, 256, 0, stream>>>(mem_h,   WtK, bk, Kb, vocab);
    gemm256_bf16<<<(vocab + 127) / 128, 256, 0, stream>>>(mem_l,   WtV, bv, Vb, vocab);

    // 3) gather + softmax attention
    float* outh = (float*)d_out;
    float* outl = outh + (size_t)num * DIM;
    int blocks = (num * 32 + 255) / 256;   // one wave per token
    attn_kernel<<<blocks, 256, 0, stream>>>(Qb, Kb, Vb, idx, def_h, def_l,
                                            outh, outl, num);
}
